// xLSTM_43550968381546
// MI455X (gfx1250) — matrix operand order repaired and checked by reading, not compiled, not run on hardware
//
#include <hip/hip_runtime.h>
#include <hip/hip_bf16.h>

// ---------------------------------------------------------------------------
// xLSTM forward for MI455X (gfx1250, wave32, WMMA bf16, async global->LDS)
// B=16, S=1024, D_IN=512, H=1024, NH=4, HD=256
// ---------------------------------------------------------------------------

#define USE_ASYNC_LDS 1

typedef __attribute__((ext_vector_type(16))) __bf16 v16bf;
typedef __attribute__((ext_vector_type(8)))  __bf16 v8bf;
typedef __attribute__((ext_vector_type(8)))  float  v8f;

union BF16x16 { v16bf v; v8bf h[2]; };

__device__ inline v8f zero_v8f() {
    v8f z;
#pragma unroll
    for (int i = 0; i < 8; ++i) z[i] = 0.0f;
    return z;
}

__device__ inline v8f wmma_bf16(v16bf a, v16bf b, v8f c) {
    // (neg_a, A, neg_b, B, c_mod, C, reuse_a, reuse_b)
    return __builtin_amdgcn_wmma_f32_16x16x32_bf16(false, a, false, b,
                                                   (short)0, c, false, false);
}

// A-fragment (16x32 bf16): lane m = lane%16.
// lane<16: K = {0..7, 16..23}; lane>=16: K = {8..15, 24..31}.
__device__ inline v16bf load_a_bf16(const __bf16* rowptr, int khalfA) {
    BF16x16 f;
    f.h[0] = *(const v8bf*)(rowptr + khalfA);
    f.h[1] = *(const v8bf*)(rowptr + 16 + khalfA);
    return f.v;
}

// B-fragment (32x16 bf16): lane n = lane%16; lane<16: K = {0..7, 16..23}; lane>=16: K = {8..15, 24..31}.
__device__ inline v16bf load_b_bf16(const __bf16* p) {
    BF16x16 f;
    f.h[0] = *(const v8bf*)(p);
    f.h[1] = *(const v8bf*)(p + 16);
    return f.v;
}

// 16-byte global -> LDS copy. Async path (ASYNCcnt-tracked) when enabled.
__device__ inline void cp16_g2l(__bf16* lds_dst, const __bf16* gsrc) {
#if USE_ASYNC_LDS
    unsigned ldsoff = (unsigned)(uintptr_t)lds_dst;   // ISA 10.2: LDS = addr[31:0]
    asm volatile("global_load_async_to_lds_b128 %0, %1, off"
                 :: "v"(ldsoff), "v"(gsrc) : "memory");
#else
    *(v8bf*)lds_dst = *(const v8bf*)gsrc;
#endif
}

__device__ inline void g2l_wait0() {
#if USE_ASYNC_LDS
    asm volatile("s_wait_asynccnt 0x0" ::: "memory");
#endif
}

// Async loads complete in-order: waiting ASYNCcnt<=3 lets the 3 copies of the
// *next* tile stay in flight while guaranteeing the previous tile has landed.
__device__ inline void g2l_wait3() {
#if USE_ASYNC_LDS
    asm volatile("s_wait_asynccnt 0x3" ::: "memory");
#endif
}

// ---------------------------------------------------------------------------
// Prep kernels: one-time f32 -> bf16 casts (+ layout changes) so all GEMM
// staging/fragments are raw 16B moves.
// ---------------------------------------------------------------------------
__global__ __launch_bounds__(256) void xcast_kernel(const float* __restrict__ x,
                                                    __bf16* __restrict__ xbf) {
    int idx = blockIdx.x * 256 + threadIdx.x;         // 16384*512
    xbf[idx] = (__bf16)x[idx];
}

// W_proj (K=512, N=1024) -> Wpt[n][k] bf16 (K-contiguous per output column)
__global__ __launch_bounds__(256) void wpt_kernel(const float* __restrict__ Wp,
                                                  __bf16* __restrict__ Wpt) {
    int idx = blockIdx.x * 256 + threadIdx.x;         // 1024*512
    int n = idx >> 9;
    int k = idx & 511;
    Wpt[idx] = (__bf16)Wp[(size_t)k * 1024 + n];
}

// Wi/Wf/Wz/Wo (NH,HD,HD) -> Wg[gate][head][o][i] bf16 (already K-contiguous)
__global__ __launch_bounds__(256) void wg_cast_kernel(const float* __restrict__ Wi,
                                                      const float* __restrict__ Wf,
                                                      const float* __restrict__ Wz,
                                                      const float* __restrict__ Wo,
                                                      __bf16* __restrict__ Wg) {
    int idx = blockIdx.x * 256 + threadIdx.x;         // 4*262144
    int g = idx >> 18;
    int rem = idx & 262143;
    const float* src = (g == 0) ? Wi : (g == 1) ? Wf : (g == 2) ? Wz : Wo;
    Wg[idx] = (__bf16)src[rem];
}

// R (NH,HD,4H) f32 -> Rt[n][k][i] bf16 (K=i contiguous per output column k)
__global__ __launch_bounds__(256) void rt_prep_kernel(const float* __restrict__ R,
                                                      __bf16* __restrict__ Rt) {
    int idx = blockIdx.x * 256 + threadIdx.x;         // 4*1024*256
    int n = idx >> 18;
    int rem = idx & ((1 << 18) - 1);
    int k = rem >> 8;
    int i = rem & 255;
    Rt[idx] = (__bf16)R[((size_t)n * 256 + i) * 1024 + k];
}

// ---------------------------------------------------------------------------
// Kernel 1: x_proj = x @ W_proj + b_proj   (M=16384, K=512, N=1024)
// block = 256 thr (8 waves), tile 64(M) x 128(N); double-buffered async
// global->LDS staging overlapped with the WMMA chain.
// ---------------------------------------------------------------------------
__global__ __launch_bounds__(256) void proj_gemm_kernel(const __bf16* __restrict__ xbf,
                                                        const __bf16* __restrict__ Wpt,
                                                        const float* __restrict__ bp,
                                                        float* __restrict__ xproj) {
    __shared__ __bf16 Al[2][64][32];     // [buf][m][k]
    __shared__ __bf16 Bl[2][128][32];    // [buf][n][k]

    const int tid  = threadIdx.x;
    const int lane = tid & 31;
    const int wave = tid >> 5;
    const int m0 = blockIdx.x * 64;
    const int n0 = blockIdx.y * 128;
    const int wm = wave & 3;          // M sub-tile
    const int wn = wave >> 2;         // N sub-tile (64 wide)
    const int lm = lane & 15;
    const int hi = lane >> 4;
    const int khalfA = hi ? 8 : 0;
    const int khalfB = hi ? 8 : 0;

    // per-thread staging coordinates (3 x 16B chunks per tile)
    const int rowA = tid >> 2;
    const int offA = (tid & 3) * 8;
    const int rowB0 = (tid * 2) >> 2;
    const int offB0 = ((tid * 2) & 3) * 8;
    const int rowB1 = (tid * 2 + 1) >> 2;
    const int offB1 = ((tid * 2 + 1) & 3) * 8;

    v8f acc[4];
#pragma unroll
    for (int j = 0; j < 4; ++j) acc[j] = zero_v8f();

    auto stage = [&](int buf, int k0) {
        cp16_g2l(&Al[buf][rowA][offA],  xbf + (size_t)(m0 + rowA) * 512 + k0 + offA);
        cp16_g2l(&Bl[buf][rowB0][offB0], Wpt + (size_t)(n0 + rowB0) * 512 + k0 + offB0);
        cp16_g2l(&Bl[buf][rowB1][offB1], Wpt + (size_t)(n0 + rowB1) * 512 + k0 + offB1);
    };

    stage(0, 0);
    g2l_wait0();
    __syncthreads();

    for (int kt = 0; kt < 16; ++kt) {
        int buf = kt & 1;
        if (kt + 1 < 16)
            stage(buf ^ 1, (kt + 1) * 32);   // prefetch next tile (stays in flight)

        v16bf a = load_a_bf16(&Al[buf][wm * 16 + lm][0], khalfA);
#pragma unroll
        for (int j = 0; j < 4; ++j) {
            v16bf b = load_b_bf16(&Bl[buf][wn * 64 + j * 16 + lm][khalfB]);
            acc[j] = wmma_bf16(a, b, acc[j]);
        }
        g2l_wait3();       // previous tile's copies (all waves) have landed
        __syncthreads();   // all waves done reading buf; next iter may overwrite it
    }

    // epilogue: D[m][n], M = r + 8*hi, N = lm
#pragma unroll
    for (int j = 0; j < 4; ++j) {
        int n = n0 + wn * 64 + j * 16 + lm;
        float bias = bp[n];
#pragma unroll
        for (int r = 0; r < 8; ++r) {
            int m = m0 + wm * 16 + r + 8 * hi;
            xproj[(size_t)m * 1024 + n] = acc[j][r] + bias;
        }
    }
}

// ---------------------------------------------------------------------------
// Kernel 2: pre = LayerNorm(x_proj) * ln_w  -> bf16
// ---------------------------------------------------------------------------
__global__ __launch_bounds__(256) void ln_kernel(const float* __restrict__ xproj,
                                                 const float* __restrict__ lnw,
                                                 __bf16* __restrict__ pre) {
    __shared__ float red[2][256];
    const int row = blockIdx.x;
    const int tid = threadIdx.x;
    const float* xr = xproj + (size_t)row * 1024;

    float v[4], s = 0.f, s2 = 0.f;
#pragma unroll
    for (int j = 0; j < 4; ++j) {
        v[j] = xr[tid + j * 256];
        s += v[j];
        s2 += v[j] * v[j];
    }
    red[0][tid] = s; red[1][tid] = s2;
    __syncthreads();
    for (int off = 128; off > 0; off >>= 1) {
        if (tid < off) {
            red[0][tid] += red[0][tid + off];
            red[1][tid] += red[1][tid + off];
        }
        __syncthreads();
    }
    float mu  = red[0][0] * (1.f / 1024.f);
    float var = red[1][0] * (1.f / 1024.f) - mu * mu;
    float inv = rsqrtf(var + 1e-5f);
#pragma unroll
    for (int j = 0; j < 4; ++j) {
        int h = tid + j * 256;
        pre[(size_t)row * 1024 + h] = (__bf16)((v[j] - mu) * inv * lnw[h]);
    }
}

// ---------------------------------------------------------------------------
// Kernel 3: gate GEMMs. y[m,o] = sum_i pre[m, n*HD+i] * Wg[n][o][i] + bg
// gates laid out (S, 4, H, B) f32 so the scan reads contiguous batch vectors.
// grid (1024, 8), block 256 = 8 waves; each wave one 16x64 tile.
// All operands bf16, fragments are direct 16B global loads (L2-resident).
// ---------------------------------------------------------------------------
__global__ __launch_bounds__(256) void gates_gemm_kernel(const __bf16* __restrict__ pre,
                                                         const __bf16* __restrict__ Wg,
                                                         const float* __restrict__ bi,
                                                         const float* __restrict__ bfv,
                                                         const float* __restrict__ bz,
                                                         const float* __restrict__ bo,
                                                         float* __restrict__ gates) {
    const int tid  = threadIdx.x;
    const int lane = tid & 31;
    const int wave = tid >> 5;
    const int m0 = blockIdx.x * 16;
    const int wg = blockIdx.y * 8 + wave;   // 0..63
    const int hg = wg >> 2;                 // head-gate 0..15
    const int nsub = wg & 3;                // 64-wide N sub-tile
    const int head = hg >> 2;
    const int gate = hg & 3;
    const float* bias = (gate == 0) ? bi : (gate == 1) ? bfv : (gate == 2) ? bz : bo;

    const int lm = lane & 15;
    const int hi = lane >> 4;
    const int khalfA = hi ? 8 : 0;
    const int khalfB = hi ? 8 : 0;

    v8f acc[4];
#pragma unroll
    for (int j = 0; j < 4; ++j) acc[j] = zero_v8f();

    const __bf16* arow = pre + (size_t)(m0 + lm) * 1024 + head * 256;
    const __bf16* wbase = Wg + ((size_t)gate * 4 + head) * 65536;   // [o][i]
    for (int k0 = 0; k0 < 256; k0 += 32) {
        v16bf a = load_a_bf16(arow + k0, khalfA);
#pragma unroll
        for (int j = 0; j < 4; ++j) {
            int o = nsub * 64 + j * 16 + lm;
            v16bf b = load_b_bf16(wbase + (size_t)o * 256 + k0 + khalfB);
            acc[j] = wmma_bf16(a, b, acc[j]);
        }
    }
#pragma unroll
    for (int j = 0; j < 4; ++j) {
        int o = nsub * 64 + j * 16 + lm;
        int h = head * 256 + o;
        float bv = bias[h];
#pragma unroll
        for (int r = 0; r < 8; ++r) {
            int m = m0 + r + 8 * hi;
            int b = m >> 10;       // m = b*S + s, S=1024
            int s = m & 1023;
            gates[(((size_t)s * 4 + gate) * 1024 + h) * 16 + b] = acc[j][r] + bv;
        }
    }
}

// ---------------------------------------------------------------------------
// Kernel 4: sequential xLSTM scan. One workgroup per head (grid=4, 512 thr,
// 16 waves). h (16 x 256) lives in LDS as bf16 A-matrix; each wave owns 16
// head-columns and computes all 4 gate recurrences for them (B=16 = one WMMA
// M tile). States c/n/m in LDS for all 1024 steps. 32 WMMAs / wave / step.
// ---------------------------------------------------------------------------
__global__ __launch_bounds__(512) void scan_kernel(const float* __restrict__ gates,
                                                   const __bf16* __restrict__ Rt,
                                                   const float* __restrict__ rb,
                                                   float* __restrict__ ys) {
    __shared__ __bf16 hbf[16][256];
    __shared__ float cS[16][256];
    __shared__ float nS[16][256];
    __shared__ float mS[16][256];

    const int head = blockIdx.x;
    const int tid  = threadIdx.x;
    const int lane = tid & 31;
    const int wave = tid >> 5;          // 0..15

    for (int i = tid; i < 16 * 256; i += 512) {
        ((__bf16*)hbf)[i] = (__bf16)0.0f;
        ((float*)cS)[i] = 0.0f;
        ((float*)nS)[i] = 0.0f;
        ((float*)mS)[i] = 0.0f;
    }
    __syncthreads();

    const int lm = lane & 15;
    const int hi = lane >> 4;
    const int khalfA = hi ? 8 : 0;
    const int khalfB = hi ? 8 : 0;
    const int col = wave * 16 + lm;     // column within head, 0..255
    const int h   = head * 256 + col;   // global hidden index

    float rbv[4];
#pragma unroll
    for (int g = 0; g < 4; ++g) rbv[g] = rb[g * 1024 + h];

    const __bf16* RtH = Rt + (size_t)head * 1024 * 256;

    for (int t = 0; t < 1024; ++t) {
        // prefetch next step's gate slab while we do the matmul
        if (t + 1 < 1024)
            __builtin_prefetch(gates + (((size_t)(t + 1) * 4) * 1024 + h) * 16, 0, 3);

        v8f acc[4];
#pragma unroll
        for (int g = 0; g < 4; ++g) acc[g] = zero_v8f();

#pragma unroll
        for (int kk = 0; kk < 8; ++kk) {
            int i0 = kk * 32;
            v16bf a = load_a_bf16(&hbf[lm][i0], khalfA);   // A[b=lm][i]
#pragma unroll
            for (int g = 0; g < 4; ++g) {
                int krow = g * 256 + wave * 16 + lm;        // output column in 4H
                v16bf b = load_b_bf16(RtH + (size_t)krow * 256 + i0 + khalfB);
                acc[g] = wmma_bf16(a, b, acc[g]);
            }
        }
        __syncthreads();   // all waves done reading hbf

        float raw[4][8];
#pragma unroll
        for (int g = 0; g < 4; ++g) {
            const float* gp = gates + (((size_t)t * 4 + g) * 1024 + h) * 16 + hi * 8;
            float4 g0 = *(const float4*)(gp);
            float4 g1 = *(const float4*)(gp + 4);
            raw[g][0] = acc[g][0] + g0.x + rbv[g];
            raw[g][1] = acc[g][1] + g0.y + rbv[g];
            raw[g][2] = acc[g][2] + g0.z + rbv[g];
            raw[g][3] = acc[g][3] + g0.w + rbv[g];
            raw[g][4] = acc[g][4] + g1.x + rbv[g];
            raw[g][5] = acc[g][5] + g1.y + rbv[g];
            raw[g][6] = acc[g][6] + g1.z + rbv[g];
            raw[g][7] = acc[g][7] + g1.w + rbv[g];
        }

#pragma unroll
        for (int r = 0; r < 8; ++r) {
            int b = r + 8 * hi;                 // D-tile row = batch index
            float iraw = raw[0][r], fraw = raw[1][r];
            float zraw = raw[2][r], oraw = raw[3][r];
            float c_old = cS[b][col];
            float n_old = nS[b][col];
            float m_old = mS[b][col];
            // numerically stable log-sigmoid
            float lgf = (fraw >= 0.f) ? -log1pf(expf(-fraw))
                                      : (fraw - log1pf(expf(fraw)));
            float m_new = fmaxf(iraw, m_old + lgf);
            float ig = expf(iraw - m_new);
            float fg = expf(m_old + lgf - m_new);
            float c_new = fg * c_old + ig * tanhf(zraw);
            float n_new = fg * n_old + ig;
            float og = 1.f / (1.f + expf(-oraw));
            float h_new = og * c_new / n_new;
            cS[b][col] = c_new;
            nS[b][col] = n_new;
            mS[b][col] = m_new;
            hbf[b][col] = (__bf16)h_new;
            ys[((size_t)b * 1024 + t) * 1024 + h] = h_new;   // (B,S,H)
        }
        __syncthreads();   // hbf fully updated before next step's A frags
    }
}

// ---------------------------------------------------------------------------
// Kernel 5: per-head groupnorm * gn_w, residual with x_proj, final LN * post_w
// ---------------------------------------------------------------------------
__global__ __launch_bounds__(256) void postnorm_kernel(const float* __restrict__ ys,
                                                       const float* __restrict__ xproj,
                                                       const float* __restrict__ gnw,
                                                       const float* __restrict__ postw,
                                                       float* __restrict__ out) {
    __shared__ float red[2][256];
    __shared__ float stats[4][2];
    const int row = blockIdx.x;            // b*S + s
    const int tid = threadIdx.x;
    const int headT = tid >> 6;            // 4 groups of 64 threads
    const int l64 = tid & 63;
    const float* yr = ys + (size_t)row * 1024;

    float v[4], s = 0.f, s2 = 0.f;
#pragma unroll
    for (int j = 0; j < 4; ++j) {
        int h = headT * 256 + l64 + j * 64;
        v[j] = yr[h];
        s += v[j];
        s2 += v[j] * v[j];
    }
    red[0][tid] = s; red[1][tid] = s2;
    __syncthreads();
    for (int off = 32; off > 0; off >>= 1) {
        if (l64 < off) {
            red[0][tid] += red[0][tid + off];
            red[1][tid] += red[1][tid + off];
        }
        __syncthreads();
    }
    if (l64 == 0) {
        float mu  = red[0][tid] * (1.f / 256.f);
        float var = red[1][tid] * (1.f / 256.f) - mu * mu;
        stats[headT][0] = mu;
        stats[headT][1] = rsqrtf(var + 1e-5f);
    }
    __syncthreads();

    float w[4], S1 = 0.f, S2 = 0.f;
#pragma unroll
    for (int j = 0; j < 4; ++j) {
        int h = headT * 256 + l64 + j * 64;
        float yn = (v[j] - stats[headT][0]) * stats[headT][1] * gnw[h];
        float o  = xproj[(size_t)row * 1024 + h] + yn;
        w[j] = o;
        S1 += o;
        S2 += o * o;
    }
    __syncthreads();
    red[0][tid] = S1; red[1][tid] = S2;
    __syncthreads();
    for (int off = 128; off > 0; off >>= 1) {
        if (tid < off) {
            red[0][tid] += red[0][tid + off];
            red[1][tid] += red[1][tid + off];
        }
        __syncthreads();
    }
    float mu2  = red[0][0] * (1.f / 1024.f);
    float var2 = red[1][0] * (1.f / 1024.f) - mu2 * mu2;
    float inv2 = rsqrtf(var2 + 1e-5f);
#pragma unroll
    for (int j = 0; j < 4; ++j) {
        int h = headT * 256 + l64 + j * 64;
        out[(size_t)row * 1024 + h] = (w[j] - mu2) * inv2 * postw[h];
    }
}

// ---------------------------------------------------------------------------
// launch
// ---------------------------------------------------------------------------
extern "C" void kernel_launch(void* const* d_in, const int* in_sizes, int n_in,
                              void* d_out, int out_size, void* d_ws, size_t ws_size,
                              hipStream_t stream) {
    const float* x     = (const float*)d_in[0];
    const float* Wp    = (const float*)d_in[1];
    const float* bp    = (const float*)d_in[2];
    const float* lnw   = (const float*)d_in[3];
    const float* Wi    = (const float*)d_in[4];
    const float* bi    = (const float*)d_in[5];
    const float* Wf    = (const float*)d_in[6];
    const float* bfv   = (const float*)d_in[7];
    const float* Wz    = (const float*)d_in[8];
    const float* bz    = (const float*)d_in[9];
    const float* Wo    = (const float*)d_in[10];
    const float* bo    = (const float*)d_in[11];
    const float* R     = (const float*)d_in[12];
    const float* rb    = (const float*)d_in[13];
    const float* gnw   = (const float*)d_in[14];
    const float* postw = (const float*)d_in[15];
    float* out = (float*)d_out;

    char* ws = (char*)d_ws;
    float*  xproj = (float*)(ws + 0);                 //  64 MB (16384 x 1024 f32)
    __bf16* pre   = (__bf16*)(ws + 67108864);         //  32 MB (16384 x 1024 bf16)
    float*  gates = (float*)(ws + 100663296);         // 256 MB (S,4,H,B f32)
    __bf16* Rt    = (__bf16*)(ws + 369098752);        //   2 MB (4,1024,256 bf16)
    float*  ys    = (float*)(ws + 371195904);         //  64 MB (B,S,H f32)
    __bf16* xbf   = (__bf16*)(ws + 438304768);        //  16 MB (16384 x 512 bf16)
    __bf16* Wpt   = (__bf16*)(ws + 455081984);        //   1 MB (1024 x 512 bf16)
    __bf16* Wg    = (__bf16*)(ws + 456130560);        //   2 MB (4,4,256,256 bf16)

    xcast_kernel<<<32768, 256, 0, stream>>>(x, xbf);
    wpt_kernel<<<2048, 256, 0, stream>>>(Wp, Wpt);
    wg_cast_kernel<<<4096, 256, 0, stream>>>(Wi, Wf, Wz, Wo, Wg);
    rt_prep_kernel<<<4096, 256, 0, stream>>>(R, Rt);

    proj_gemm_kernel<<<dim3(256, 8), 256, 0, stream>>>(xbf, Wpt, bp, xproj);
    ln_kernel<<<16384, 256, 0, stream>>>(xproj, lnw, pre);
    gates_gemm_kernel<<<dim3(1024, 8), 256, 0, stream>>>(pre, Wg, bi, bfv, bz, bo, gates);
    scan_kernel<<<4, 512, 0, stream>>>(gates, Rt, rb, ys);
    postnorm_kernel<<<16384, 256, 0, stream>>>(ys, xproj, gnw, postw, out);
}
